// WHGCN_Conv_27453430956156
// MI455X (gfx1250) — matrix operand order repaired and checked by reading, not compiled, not run on hardware
//
#include <hip/hip_runtime.h>
#include <hip/hip_bf16.h>

#define NN 4096
#define FF 512
#define KTT 64          // GEMM K-tile
#define ALD 72          // LDS leading dim for A tiles (bank-stagger pad)
#define BLD 72          // LDS leading dim for B tiles
#define ABSTR 520       // cos A-band LDS stride (halves)
#define CH 16           // n-chunks for prep_t partial sums
#define COS_THR 0.5f
#define COS_K 10
#define SCALERV 1.0f
#define EPSF 1e-12f

typedef __attribute__((ext_vector_type(16))) _Float16 v16h;
typedef __attribute__((ext_vector_type(8)))  float    v8f;
typedef int v4i_vs __attribute__((vector_size(16)));   // matches builtin param pointee

#if defined(__has_builtin)
#if __has_builtin(__builtin_amdgcn_global_load_async_to_lds_b128)
#define HAVE_ASYNC_LDS 1
#endif
#endif

// ---- WMMA fragment loaders (16x16x32 f16, contiguous k = 16*(lane/16) + s in A and B) ----
static __device__ __forceinline__ v16h load_a_frag(const _Float16* base, int ld, int row0, int k0) {
  int l   = threadIdx.x & 31;
  int row = row0 + (l & 15);
  int hi  = (l >> 4) & 1;
  const _Float16* p0 = base + (size_t)row * ld + k0 + (hi ? 16 : 0);
  const _Float16* p1 = base + (size_t)row * ld + k0 + (hi ? 24 : 8);
  v16h f;
#pragma unroll
  for (int i = 0; i < 8; ++i) f[i] = p0[i];
#pragma unroll
  for (int i = 0; i < 8; ++i) f[8 + i] = p1[i];
  return f;
}

static __device__ __forceinline__ v16h load_b_frag(const _Float16* base, int ld, int col0, int k0) {
  int l    = threadIdx.x & 31;
  int col  = col0 + (l & 15);
  int koff = (l >> 4) ? 16 : 0;
  const _Float16* p = base + (size_t)col * ld + k0 + koff;
  v16h f;
#pragma unroll
  for (int i = 0; i < 16; ++i) f[i] = p[i];
  return f;
}

// ---- 1) row L2-normalize x -> xn16, raw x -> x16 ----
__global__ void k_rownorm(const float* __restrict__ x, _Float16* __restrict__ xn16,
                          _Float16* __restrict__ x16) {
  __shared__ float red[128];
  int row = blockIdx.x;
  int t   = threadIdx.x;
  const float* xr = x + (size_t)row * FF;
  float v[4];
  float s = 0.f;
#pragma unroll
  for (int i = 0; i < 4; ++i) { v[i] = xr[t + i * 128]; s += v[i] * v[i]; }
  red[t] = s;
  __syncthreads();
  for (int off = 64; off > 0; off >>= 1) {
    if (t < off) red[t] += red[t + off];
    __syncthreads();
  }
  float scale = 1.f / fmaxf(sqrtf(red[0]), EPSF);
#pragma unroll
  for (int i = 0; i < 4; ++i) {
    xn16[(size_t)row * FF + t + i * 128] = (_Float16)(v[i] * scale);
    x16 [(size_t)row * FF + t + i * 128] = (_Float16)v[i];
  }
}

// ---- 2) fp32 -> fp16 convert (proj_w) ----
__global__ void k_cvt16(const float* __restrict__ src, _Float16* __restrict__ dst, int n) {
  int i = blockIdx.x * blockDim.x + threadIdx.x;
  if (i < n) dst[i] = (_Float16)src[i];
}

// ---- 3) projection: h = x @ proj_w^T + b ----
__global__ __launch_bounds__(128) void k_proj(const _Float16* __restrict__ x16,
                                              const _Float16* __restrict__ pw16,
                                              const float* __restrict__ bias,
                                              _Float16* __restrict__ h16) {
  int wave = threadIdx.x >> 5;
  int r0   = blockIdx.x * 16;
  int c0   = blockIdx.y * 64 + wave * 16;
  v8f acc = {};
#pragma unroll
  for (int k0 = 0; k0 < FF; k0 += 32) {
    v16h a = load_a_frag(x16, FF, r0, k0);
    v16h b = load_b_frag(pw16, FF, c0, k0);
    acc = __builtin_amdgcn_wmma_f32_16x16x32_f16(false, a, false, b, (short)0, acc, false, false);
  }
  int l   = threadIdx.x & 31;
  int col = c0 + (l & 15);
  float bb = bias[col];
#pragma unroll
  for (int i = 0; i < 8; ++i) {
    int m = (l < 16) ? i : i + 8;
    h16[(size_t)(r0 + m) * FF + col] = (_Float16)(acc[i] + bb);
  }
}

// ---- 4) cos band in 256KB LDS -> top-10 -> mask bits ----
__global__ __launch_bounds__(256) void k_cos_mask(const _Float16* __restrict__ xn16,
                                                  unsigned* __restrict__ maskbits) {
  extern __shared__ float cosb[];                        // 16*NN f32
  _Float16* aband = (_Float16*)(cosb + 16 * NN);         // 16 x ABSTR halves
  float* kthv = (float*)(aband + 16 * ABSTR);            // 16 f32
  int wave = threadIdx.x >> 5;
  int l    = threadIdx.x & 31;
  int r0   = blockIdx.x * 16;

  // stage the 16x512 A band into LDS (16B chunks per thread), async if available
#pragma unroll
  for (int j = 0; j < 4; ++j) {
    int c   = threadIdx.x + j * 256;       // 1024 chunks of 8 halves
    int row = c >> 6;
    int off = (c & 63) * 8;
#ifdef HAVE_ASYNC_LDS
    __builtin_amdgcn_global_load_async_to_lds_b128(
        (v4i_vs*)(const void*)(xn16 + (size_t)(r0 + row) * FF + off),
        (v4i_vs*)(void*)(aband + row * ABSTR + off),
        0, 0);
#else
#pragma unroll
    for (int i = 0; i < 8; ++i)
      aband[row * ABSTR + off + i] = xn16[(size_t)(r0 + row) * FF + off + i];
#endif
  }
#ifdef HAVE_ASYNC_LDS
#if __has_builtin(__builtin_amdgcn_s_wait_asynccnt)
  __builtin_amdgcn_s_wait_asynccnt(0);
#else
  asm volatile("s_wait_asynccnt 0x0" ::: "memory");
#endif
#endif
  __syncthreads();

  // phase 1: WMMA full 16 x 4096 cos band into LDS
  for (int jt = wave; jt < NN / 16; jt += 8) {
    int c0 = jt * 16;
    v8f acc = {};
#pragma unroll
    for (int k0 = 0; k0 < FF; k0 += 32) {
      v16h a = load_a_frag(aband, ABSTR, 0, k0);
      v16h b = load_b_frag(xn16, FF, c0, k0);
      acc = __builtin_amdgcn_wmma_f32_16x16x32_f16(false, a, false, b, (short)0, acc, false, false);
    }
#pragma unroll
    for (int i = 0; i < 8; ++i) {
      int m = (l < 16) ? i : i + 8;
      cosb[m * NN + c0 + (l & 15)] = acc[i];
    }
  }
  __syncthreads();

  // phase 2: 10th-largest per row (register-resident compare-exchange chains)
  for (int rr = 0; rr < 2; ++rr) {
    int r = wave * 2 + rr;
    float t[COS_K];
#pragma unroll
    for (int q = 0; q < COS_K; ++q) t[q] = -3.0e38f;
    const float* rowp = cosb + (size_t)r * NN;
    for (int j = l; j < NN; j += 32) {
      float v = rowp[j];
#pragma unroll
      for (int q = 0; q < COS_K; ++q) {        // sorted insert, constant idx
        float mx = fmaxf(t[q], v);
        v = fminf(t[q], v);
        t[q] = mx;
      }
    }
    float kth = -3.0e38f;
#pragma unroll
    for (int it = 0; it < COS_K; ++it) {       // extract 10 wave-wide maxima
      float cand = t[0];
      float m = cand;
#pragma unroll
      for (int off = 16; off > 0; off >>= 1) m = fmaxf(m, __shfl_xor(m, off));
      unsigned long long b = __ballot(cand == m);
      int wl = (int)__ffsll(b) - 1;
      if (l == wl) {
#pragma unroll
        for (int q = 0; q < COS_K - 1; ++q) t[q] = t[q + 1];
        t[COS_K - 1] = -3.0e38f;
      }
      kth = m;
    }
    if (l == 0) kthv[r] = kth;
  }
  __syncthreads();

  // phase 3: mask bits
  for (int rr = 0; rr < 2; ++rr) {
    int r = wave * 2 + rr;
    float kth = kthv[r];
    const float* rowp = cosb + (size_t)r * NN;
    for (int w = 0; w < NN / 32; ++w) {
      float v = rowp[w * 32 + l];
      int pred = (v > COS_THR) || (v >= kth);
      unsigned long long b = __ballot(pred);
      if (l == 0) maskbits[(size_t)(r0 + r) * (NN / 32) + w] = (unsigned)b;
    }
  }
}

// ---- 5) materialize w1m16 = fp16(w1*H*mask), partial column L1 sums ----
__global__ void k_prep_t(const float* __restrict__ w1, const float* __restrict__ Hm,
                         const unsigned* __restrict__ mb, _Float16* __restrict__ w1m16,
                         float* __restrict__ part) {
  int e = blockIdx.x * 256 + threadIdx.x;
  int n0 = blockIdx.y * (NN / CH);
  unsigned sh = e & 31;
  int word = e >> 5;
  float s = 0.f;
  for (int n = n0; n < n0 + NN / CH; ++n) {
    unsigned w = mb[(size_t)n * (NN / 32) + word];
    float v = ((w >> sh) & 1u) ? w1[(size_t)n * NN + e] * Hm[(size_t)n * NN + e] : 0.f;
    w1m16[(size_t)n * NN + e] = (_Float16)v;
    s += fabsf(v);
  }
  part[(size_t)blockIdx.y * NN + e] = s;
}

__global__ void k_red(const float* __restrict__ part, float* __restrict__ cs) {
  int e = blockIdx.x * 256 + threadIdx.x;
  float s = 0.f;
#pragma unroll
  for (int c = 0; c < CH; ++c) s += part[(size_t)c * NN + e];
  cs[e] = s;
}

// ---- 6) materialize w2m16 = fp16(w2*H*mask), row L1 sums (deterministic) ----
__global__ void k_prep_n(const float* __restrict__ w2, const float* __restrict__ Hm,
                         const unsigned* __restrict__ mb, _Float16* __restrict__ w2m16,
                         float* __restrict__ rs) {
  __shared__ float red[256];
  int t = threadIdx.x;
  for (int r = 0; r < 8; ++r) {
    int n = blockIdx.x * 8 + r;
    __syncthreads();
    float s = 0.f;
    for (int e = t; e < NN; e += 256) {
      unsigned w = mb[(size_t)n * (NN / 32) + (e >> 5)];
      float v = ((w >> (e & 31)) & 1u) ? w2[(size_t)n * NN + e] * Hm[(size_t)n * NN + e] : 0.f;
      w2m16[(size_t)n * NN + e] = (_Float16)v;
      s += fabsf(v);
    }
    red[t] = s;
    __syncthreads();
    for (int off = 128; off > 0; off >>= 1) {
      if (t < off) red[t] += red[t + off];
      __syncthreads();
    }
    if (t == 0) rs[n] = red[0];
  }
}

// ---- 7) plain fp16 GEMM 64x128 tile, row-scaled epilogue ----
// TRANS: A element (r, k) = A16[k*NN + r] (transposed read), else A16[r*NN + k]
template <bool TRANS>
__global__ __launch_bounds__(256) void k_gemm(const _Float16* __restrict__ A16,
                                              const _Float16* __restrict__ B16,
                                              const float* __restrict__ scl,
                                              _Float16* __restrict__ out16,
                                              float* __restrict__ out32) {
  __shared__ _Float16 lab[64 * ALD];
  __shared__ _Float16 lbt[128 * BLD];
  int wave = threadIdx.x >> 5;
  int l    = threadIdx.x & 31;
  int r0   = blockIdx.x * 64;
  int col0 = blockIdx.y * 128;
  v8f acc[4] = {};
  for (int k0 = 0; k0 < NN; k0 += KTT) {
    if (k0 + KTT < NN) {
      __builtin_prefetch(&A16[TRANS ? (size_t)(k0 + KTT) * NN + r0
                                    : (size_t)r0 * NN + k0 + KTT], 0, 1);
      __builtin_prefetch(&B16[(size_t)(k0 + KTT) * FF + col0], 0, 1);
    }
#pragma unroll
    for (int j = 0; j < (64 * KTT) / 256; ++j) {     // stage A 64xKTT
      int idx = threadIdx.x + j * 256;
      int m, kk;
      if (TRANS) { m = idx & 63; kk = idx >> 6; }
      else       { kk = idx & 63; m = idx >> 6; }
      _Float16 v = TRANS ? A16[(size_t)(k0 + kk) * NN + r0 + m]
                         : A16[(size_t)(r0 + m) * NN + k0 + kk];
      lab[m * ALD + kk] = v;
    }
#pragma unroll
    for (int j = 0; j < (128 * KTT) / 256; ++j) {    // stage B KTTx128 (transposed)
      int idx = threadIdx.x + j * 256;
      int col = idx & 127, kk = idx >> 7;
      lbt[col * BLD + kk] = B16[(size_t)(k0 + kk) * FF + col0 + col];
    }
    __syncthreads();
#pragma unroll
    for (int kk0 = 0; kk0 < KTT; kk0 += 32) {
      v16h b = load_b_frag(lbt, BLD, wave * 16, kk0);
#pragma unroll
      for (int mt = 0; mt < 4; ++mt) {
        v16h a = load_a_frag(lab, ALD, mt * 16, kk0);
        acc[mt] = __builtin_amdgcn_wmma_f32_16x16x32_f16(false, a, false, b, (short)0,
                                                         acc[mt], false, false);
      }
    }
    __syncthreads();
  }
  int colw = col0 + wave * 16 + (l & 15);
#pragma unroll
  for (int mt = 0; mt < 4; ++mt) {
#pragma unroll
    for (int i = 0; i < 8; ++i) {
      int m = (l < 16) ? i : i + 8;
      int r = r0 + mt * 16 + m;
      float v = acc[mt][i] * (SCALERV / fmaxf(scl[r], EPSF));
      if (out16) out16[(size_t)r * FF + colw] = (_Float16)v;
      else       out32[(size_t)r * FF + colw] = v;
    }
  }
}

extern "C" void kernel_launch(void* const* d_in, const int* in_sizes, int n_in,
                              void* d_out, int out_size, void* d_ws, size_t ws_size,
                              hipStream_t stream) {
  const float* x  = (const float*)d_in[0];
  const float* Hm = (const float*)d_in[1];
  const float* pw = (const float*)d_in[2];
  const float* pb = (const float*)d_in[3];
  const float* w1 = (const float*)d_in[4];
  const float* w2 = (const float*)d_in[5];
  float* out = (float*)d_out;

  char* ws = (char*)d_ws;
  _Float16* xn16  = (_Float16*)(ws);                            // 4 MB
  _Float16* x16   = (_Float16*)(ws + ((size_t)4 << 20));        // 4 MB
  _Float16* pw16  = (_Float16*)(ws + ((size_t)8 << 20));        // 0.5 MB
  _Float16* h16   = (_Float16*)(ws + ((size_t)9 << 20));        // 4 MB
  _Float16* h2    = (_Float16*)(ws + ((size_t)13 << 20));       // 4 MB
  unsigned* mb    = (unsigned*)(ws + ((size_t)17 << 20));       // 2 MB
  float*    cs1   = (float*)(ws + ((size_t)19 << 20));          // 16 KB
  float*    rs2   = (float*)(ws + ((size_t)19 << 20) + 16384);  // 16 KB
  float*    part1 = (float*)(ws + ((size_t)19 << 20) + 65536);  // 256 KB
  _Float16* w1m16 = (_Float16*)(ws + ((size_t)20 << 20));       // 32 MB
  _Float16* w2m16 = (_Float16*)(ws + ((size_t)52 << 20));       // 32 MB

  k_rownorm<<<NN, 128, 0, stream>>>(x, xn16, x16);
  k_cvt16<<<(FF * FF + 255) / 256, 256, 0, stream>>>(pw, pw16, FF * FF);
  k_proj<<<dim3(NN / 16, FF / 64), 128, 0, stream>>>(x16, pw16, pb, h16);

  size_t cos_lds = (size_t)16 * NN * sizeof(float) + (size_t)16 * ABSTR * 2 + 64;
  k_cos_mask<<<NN / 16, 256, cos_lds, stream>>>(xn16, mb);

  k_prep_t<<<dim3(NN / 256, CH), 256, 0, stream>>>(w1, Hm, mb, w1m16, part1);
  k_red<<<NN / 256, 256, 0, stream>>>(part1, cs1);
  k_prep_n<<<NN / 8, 256, 0, stream>>>(w2, Hm, mb, w2m16, rs2);

  k_gemm<true><<<dim3(NN / 64, FF / 128), 256, 0, stream>>>(w1m16, h16, cs1, h2, nullptr);
  k_gemm<false><<<dim3(NN / 64, FF / 128), 256, 0, stream>>>(w2m16, h2, rs2, nullptr, out);
}
